// HeteroGNNModel_1099511628160
// MI455X (gfx1250) — matrix operand order repaired and checked by reading, not compiled, not run on hardware
//
#include <hip/hip_runtime.h>

// ---------------- WMMA helper types (CDNA5 wave32) ----------------
typedef __attribute__((ext_vector_type(16))) __bf16    bf16x16;
typedef __attribute__((ext_vector_type(8)))  unsigned  u32x8;
typedef __attribute__((ext_vector_type(8)))  float     f32x8;

#define D_DIM 128

// Pack two f32 into one dword of two bf16 (round-half-up) with a single v_perm_b32.
static __device__ __forceinline__ unsigned pack_bf16_pair(float lo, float hi) {
    unsigned a = __float_as_uint(lo) + 0x8000u;
    unsigned b = __float_as_uint(hi) + 0x8000u;
    return __builtin_amdgcn_perm(b, a, 0x07060302u);
}

// A-matrix 16x32 bf16 per-lane layout used here (contiguous k, same order as B):
// lanes 0-15: row M=lane, K = 0..15; lanes 16-31: row M=lane-16, K = 16..31
static __device__ __forceinline__ bf16x16 load_A_bf(const float* __restrict__ p) {
    float4 a = *(const float4*)(p);
    float4 b = *(const float4*)(p + 4);
    float4 c = *(const float4*)(p + 8);
    float4 d = *(const float4*)(p + 12);
    u32x8 r;
    r[0] = pack_bf16_pair(a.x, a.y); r[1] = pack_bf16_pair(a.z, a.w);
    r[2] = pack_bf16_pair(b.x, b.y); r[3] = pack_bf16_pair(b.z, b.w);
    r[4] = pack_bf16_pair(c.x, c.y); r[5] = pack_bf16_pair(c.z, c.w);
    r[6] = pack_bf16_pair(d.x, d.y); r[7] = pack_bf16_pair(d.z, d.w);
    return __builtin_bit_cast(bf16x16, r);
}

// B-matrix 32x16 bf16: lane n (0-15) = column N=n holding K=0..15, lane n+16 = K=16..31.
// B[k][n] = W[n][k] -> 16 contiguous bf16 (32 B, aligned) from pre-converted weight row.
static __device__ __forceinline__ bf16x16 load_B_bf16(const unsigned short* __restrict__ Wbf,
                                                      int n, int k0, int lane) {
    const int kb = (lane < 16) ? 0 : 16;
    const unsigned short* p = Wbf + (long)n * D_DIM + k0 + kb;
    u32x8 r = *(const u32x8*)p;                 // 2x global_load_b128
    return __builtin_bit_cast(bf16x16, r);
}

// ---------------- Weight f32 -> bf16 pre-conversion ----------------
__global__ __launch_bounds__(256)
void wconv_kernel(const float* __restrict__ W, unsigned short* __restrict__ Wbf, int npairs)
{
    int i = blockIdx.x * 256 + threadIdx.x;
    if (i >= npairs) return;
    float2 f = ((const float2*)W)[i];
    ((unsigned*)Wbf)[i] = pack_bf16_pair(f.x, f.y);
}

// ---------------- Fused (dual) GEMM:  Y = op( X1@W1^T + b1 [+ X2@W2^T + b2] ) ----------------
// One wave per 16-row block, all 8 column tiles (8 f32x8 accumulators).
// Per K-step: issue all 8 B-tile loads first (clause), convert A while they fly,
// then run the 8-WMMA burst.
__global__ __launch_bounds__(256)
void gemm_wmma_kernel(const float* __restrict__ X1, const unsigned short* __restrict__ W1bf,
                      const float* __restrict__ b1,
                      const float* __restrict__ X2, const unsigned short* __restrict__ W2bf,
                      const float* __restrict__ b2,
                      float* __restrict__ Y, int M, int do_relu)
{
    const int lane = threadIdx.x & 31;
    const int wave = threadIdx.x >> 5;
    const long row_tile = (long)blockIdx.x * 8 + wave;
    const int m0 = (int)(row_tile * 16);
    if (m0 >= M) return;                        // wave-uniform; EXEC stays all-1 for WMMA

    const int l15  = lane & 15;
    const int arow = m0 + l15;                  // M % 16 == 0 for NU/NI
    const int akb  = (lane < 16) ? 0 : 16;

    f32x8 acc[8] = {};

    {
        const float* xrow = X1 + (long)arow * D_DIM;
#pragma unroll
        for (int k0 = 0; k0 < D_DIM; k0 += 32) {
            bf16x16 Bt[8];
#pragma unroll
            for (int ct = 0; ct < 8; ++ct)
                Bt[ct] = load_B_bf16(W1bf, ct * 16 + l15, k0, lane);
            bf16x16 A = load_A_bf(xrow + k0 + akb);
#pragma unroll
            for (int ct = 0; ct < 8; ++ct)
                acc[ct] = __builtin_amdgcn_wmma_f32_16x16x32_bf16(false, A, false, Bt[ct],
                                                                  (short)0, acc[ct],
                                                                  false, false);
        }
    }
    if (X2) {
        const float* xrow = X2 + (long)arow * D_DIM;
#pragma unroll
        for (int k0 = 0; k0 < D_DIM; k0 += 32) {
            bf16x16 Bt[8];
#pragma unroll
            for (int ct = 0; ct < 8; ++ct)
                Bt[ct] = load_B_bf16(W2bf, ct * 16 + l15, k0, lane);
            bf16x16 A = load_A_bf(xrow + k0 + akb);
#pragma unroll
            for (int ct = 0; ct < 8; ++ct)
                acc[ct] = __builtin_amdgcn_wmma_f32_16x16x32_bf16(false, A, false, Bt[ct],
                                                                  (short)0, acc[ct],
                                                                  false, false);
        }
    }

    // C/D layout: VGPR r -> row m0 + r (lanes 0-15) or m0 + 8 + r (lanes 16-31), col = n
    const int rbase = m0 + ((lane >> 4) << 3);
#pragma unroll
    for (int ct = 0; ct < 8; ++ct) {
        const int n = ct * 16 + l15;
        const float bias = b1[n] + (b2 ? b2[n] : 0.0f);
#pragma unroll
        for (int r = 0; r < 8; ++r) {
            float v = acc[ct][r] + bias;
            if (do_relu) v = fmaxf(v, 0.0f);
            Y[(long)(rbase + r) * D_DIM + n] = v;
        }
    }
}

// ---------------- Elementwise ReLU copy (float4 granularity) ----------------
__global__ __launch_bounds__(256)
void relu_copy_kernel(const float* __restrict__ x, float* __restrict__ y, long n4)
{
    long i = (long)blockIdx.x * 256 + threadIdx.x;
    if (i >= n4) return;
    float4 v = ((const float4*)x)[i];
    v.x = fmaxf(v.x, 0.0f); v.y = fmaxf(v.y, 0.0f);
    v.z = fmaxf(v.z, 0.0f); v.w = fmaxf(v.w, 0.0f);
    ((float4*)y)[i] = v;
}

// ---------------- Edge scatter: agg[dst] += xu[src] - pred[dst]; cnt[dst] += 1 ----------------
__global__ __launch_bounds__(256)
void scatter_resid_kernel(const int* __restrict__ src, const int* __restrict__ dst,
                          const float* __restrict__ xu, const float* __restrict__ pred,
                          float* __restrict__ aggsum, float* __restrict__ cnt, int E)
{
    int e = blockIdx.x * 8 + (threadIdx.x >> 5);
    if (e >= E) return;
    const int lane = threadIdx.x & 31;
    const int s = src[e], d = dst[e];
    float4 a = *(const float4*)(xu   + (long)s * D_DIM + lane * 4);
    float4 b = *(const float4*)(pred + (long)d * D_DIM + lane * 4);
    float* o = aggsum + (long)d * D_DIM + lane * 4;
    atomicAdd(o + 0, a.x - b.x);
    atomicAdd(o + 1, a.y - b.y);
    atomicAdd(o + 2, a.z - b.z);
    atomicAdd(o + 3, a.w - b.w);
    if (lane == 0) atomicAdd(cnt + d, 1.0f);
}

// ---------------- Edge scatter: agg[dst] += xi[src]; cnt[dst] += 1 ----------------
__global__ __launch_bounds__(256)
void scatter_gather_kernel(const int* __restrict__ src, const int* __restrict__ dst,
                           const float* __restrict__ xi,
                           float* __restrict__ aggsum, float* __restrict__ cnt, int E)
{
    int e = blockIdx.x * 8 + (threadIdx.x >> 5);
    if (e >= E) return;
    const int lane = threadIdx.x & 31;
    const int s = src[e], d = dst[e];
    float4 a = *(const float4*)(xi + (long)s * D_DIM + lane * 4);
    float* o = aggsum + (long)d * D_DIM + lane * 4;
    atomicAdd(o + 0, a.x);
    atomicAdd(o + 1, a.y);
    atomicAdd(o + 2, a.z);
    atomicAdd(o + 3, a.w);
    if (lane == 0) atomicAdd(cnt + d, 1.0f);
}

// ---------------- agg[n][:] /= max(cnt[n], 1) ----------------
__global__ __launch_bounds__(256)
void mean_div_kernel(float* __restrict__ agg, const float* __restrict__ cnt, int N)
{
    long i = (long)blockIdx.x * 256 + threadIdx.x;   // one float4 per thread
    long total = (long)N * (D_DIM / 4);
    if (i >= total) return;
    int n = (int)(i >> 5);
    float inv = 1.0f / fmaxf(cnt[n], 1.0f);
    float4* p = (float4*)agg + i;
    float4 v = *p;
    v.x *= inv; v.y *= inv; v.z *= inv; v.w *= inv;
    *p = v;
}

// ---------------- head: out[m] = dot(xu[m], head_W) + head_b ----------------
__global__ __launch_bounds__(256)
void head_kernel(const float* __restrict__ xu, const float* __restrict__ W,
                 const float* __restrict__ b, float* __restrict__ out, int M)
{
    int row = blockIdx.x * 8 + (threadIdx.x >> 5);
    if (row >= M) return;
    const int lane = threadIdx.x & 31;
    float4 x = *(const float4*)(xu + (long)row * D_DIM + lane * 4);
    float4 w = *(const float4*)(W + lane * 4);
    float s = x.x * w.x + x.y * w.y + x.z * w.z + x.w * w.w;
#pragma unroll
    for (int off = 16; off; off >>= 1) s += __shfl_xor(s, off, 32);
    if (lane == 0) out[row] = s + b[0];
}

// ---------------- host orchestration ----------------
extern "C" void kernel_launch(void* const* d_in, const int* in_sizes, int n_in,
                              void* d_out, int out_size, void* d_ws, size_t ws_size,
                              hipStream_t stream)
{
    const float* emb_user = (const float*)d_in[0];
    const float* emb_item = (const float*)d_in[1];
    // layer weight banks: l=0 -> inputs 2..13, l=1 -> inputs 14..25
    // j: 0 pW1, 1 pb1, 2 pW2, 3 pb2, 4 ui_Wn, 5 ui_bn, 6 ui_Ws, 7 ui_bs,
    //    8 iu_Wn, 9 iu_bn, 10 iu_Ws, 11 iu_bs
    const float* Wp[2][12];
    for (int l = 0; l < 2; ++l)
        for (int j = 0; j < 12; ++j)
            Wp[l][j] = (const float*)d_in[2 + l * 12 + j];
    const float* head_W = (const float*)d_in[26];
    const float* head_b = (const float*)d_in[27];
    const int* ue_src = (const int*)d_in[28];
    const int* ue_dst = (const int*)d_in[29];
    const int* iu_src = (const int*)d_in[30];
    const int* iu_dst = (const int*)d_in[31];

    const int NU = in_sizes[0] / D_DIM;    // 100000 (multiple of 16)
    const int NI = in_sizes[1] / D_DIM;    // 50000  (multiple of 16)
    const int E  = in_sizes[28];           // 800000

    const size_t NUb = (size_t)NU * D_DIM * sizeof(float);
    const size_t NIb = (size_t)NI * D_DIM * sizeof(float);
    const int    WN  = D_DIM * D_DIM;      // 16384 elements per weight matrix

    size_t off = 0;
    auto carve = [&](size_t bytes) -> void* {
        void* p = (void*)((char*)d_ws + off);
        off = (off + bytes + 255) & ~(size_t)255;
        return p;
    };
    float* u0 = (float*)carve(NUb); float* u1 = (float*)carve(NUb); float* u2 = (float*)carve(NUb);
    float* i0 = (float*)carve(NIb); float* i1 = (float*)carve(NIb); float* i2 = (float*)carve(NIb);
    float* cu = (float*)carve((size_t)NU * sizeof(float));
    float* ci = (float*)carve((size_t)NI * sizeof(float));
    // bf16 weight copies: per layer {pW1, pW2, uiWn, uiWs, iuWn, iuWs}
    unsigned short* Wbf[2][6];
    for (int l = 0; l < 2; ++l)
        for (int j = 0; j < 6; ++j)
            Wbf[l][j] = (unsigned short*)carve((size_t)WN * sizeof(unsigned short));
    (void)ws_size;

    dim3 blk(256);
    auto blocks1d = [](long work) { return dim3((unsigned)((work + 255) / 256)); };

    // pre-convert weights to bf16 (j index into Wp: 0,2,4,6,8,10)
    const int wj[6] = {0, 2, 4, 6, 8, 10};
    for (int l = 0; l < 2; ++l)
        for (int j = 0; j < 6; ++j)
            wconv_kernel<<<blocks1d(WN / 2), blk, 0, stream>>>(Wp[l][wj[j]], Wbf[l][j], WN / 2);

    // xu = relu(emb_user); xi = relu(emb_item)
    relu_copy_kernel<<<blocks1d((long)NU * 32), blk, 0, stream>>>(emb_user, u0, (long)NU * 32);
    relu_copy_kernel<<<blocks1d((long)NI * 32), blk, 0, stream>>>(emb_item, i0, (long)NI * 32);

    float* xu = u0; float* ufreeA = u1; float* ufreeB = u2;
    float* xi = i0; float* ifreeA = i1; float* ifreeB = i2;

    const dim3 gemmNI((unsigned)((NI / 16 + 7) / 8)), gemmNU((unsigned)((NU / 16 + 7) / 8));
    const dim3 edgeGrid((unsigned)((E + 7) / 8));

    for (int l = 0; l < 2; ++l) {
        const float *pb1 = Wp[l][1], *pb2 = Wp[l][3];
        const float *uibn = Wp[l][5], *uibs = Wp[l][7];
        const float *iubn = Wp[l][9], *iubs = Wp[l][11];
        const unsigned short *pW1b = Wbf[l][0], *pW2b = Wbf[l][1];
        const unsigned short *uiWnb = Wbf[l][2], *uiWsb = Wbf[l][3];
        const unsigned short *iuWnb = Wbf[l][4], *iuWsb = Wbf[l][5];

        // h = relu(xi @ pW1^T + pb1)            -> ifreeA
        gemm_wmma_kernel<<<gemmNI, blk, 0, stream>>>(xi, pW1b, pb1,
                                                     nullptr, nullptr, nullptr,
                                                     ifreeA, NI, 1);
        // pred = h @ pW2^T + pb2                -> ifreeB
        gemm_wmma_kernel<<<gemmNI, blk, 0, stream>>>(ifreeA, pW2b, pb2,
                                                     nullptr, nullptr, nullptr,
                                                     ifreeB, NI, 0);
        // agg_i = scatter_mean(xu[ue_src] - pred[ue_dst], ue_dst, NI)  -> ifreeA
        hipMemsetAsync(ifreeA, 0, NIb, stream);
        hipMemsetAsync(ci, 0, (size_t)NI * sizeof(float), stream);
        scatter_resid_kernel<<<edgeGrid, blk, 0, stream>>>(ue_src, ue_dst, xu, ifreeB,
                                                           ifreeA, ci, E);
        mean_div_kernel<<<blocks1d((long)NI * 32), blk, 0, stream>>>(ifreeA, ci, NI);

        // agg_u = scatter_mean(xi[iu_src], iu_dst, NU)                 -> ufreeA
        hipMemsetAsync(ufreeA, 0, NUb, stream);
        hipMemsetAsync(cu, 0, (size_t)NU * sizeof(float), stream);
        scatter_gather_kernel<<<edgeGrid, blk, 0, stream>>>(iu_src, iu_dst, xi,
                                                            ufreeA, cu, E);
        mean_div_kernel<<<blocks1d((long)NU * 32), blk, 0, stream>>>(ufreeA, cu, NU);

        // xi_new = relu(agg_i @ uiWn^T + uibn + xi @ uiWs^T + uibs)    -> ifreeB
        gemm_wmma_kernel<<<gemmNI, blk, 0, stream>>>(ifreeA, uiWnb, uibn,
                                                     xi, uiWsb, uibs,
                                                     ifreeB, NI, 1);
        // xu_new = relu(agg_u @ iuWn^T + iubn + xu @ iuWs^T + iubs)    -> ufreeB
        gemm_wmma_kernel<<<gemmNU, blk, 0, stream>>>(ufreeA, iuWnb, iubn,
                                                     xu, iuWsb, iubs,
                                                     ufreeB, NU, 1);

        // rotate buffers
        float* t;
        t = xi; xi = ifreeB; ifreeB = t;
        t = xu; xu = ufreeB; ufreeB = t;
    }

    // out = xu @ head_W^T + head_b
    head_kernel<<<dim3((unsigned)((NU + 7) / 8)), blk, 0, stream>>>(xu, head_W, head_b,
                                                                    (float*)d_out, NU);
    (void)n_in; (void)out_size;
}
